// TransformerBlock_7404523618568
// MI455X (gfx1250) — matrix operand order repaired and checked by reading, not compiled, not run on hardware
//
#include <hip/hip_runtime.h>
#include <cmath>

// ---------------- types ----------------
typedef __attribute__((ext_vector_type(16))) __bf16   v16bf;
typedef __attribute__((ext_vector_type(8)))  float    v8f;

union Frag { v16bf v; unsigned u[8]; };

#define BN  2
#define TN  2048
#define DN  768
#define HN  12
#define HDN 64
#define BT  4096      // BN*TN
#define D4  3072      // 4*DN
#define LDA 40        // LDS row stride in halves (80B, 16B-aligned, conflict-light)

__device__ __forceinline__ v8f zero8() {
  v8f r;
#pragma unroll
  for (int i = 0; i < 8; ++i) r[i] = 0.0f;
  return r;
}

__device__ __forceinline__ unsigned pack2(float a, float b) {
  union { __bf16 h[2]; unsigned u; } x;
  x.h[0] = (__bf16)a; x.h[1] = (__bf16)b;
  return x.u;
}

// A-fragment (16x32 bf16, contiguous k order, same as B): lane m = lane&15; hi = lane>>4
// v0..7 hold K = 16*hi + {2v,2v+1}
__device__ __forceinline__ void a_frag_load(const __bf16* rowp, int hi, unsigned* u) {
  const unsigned* p0 = (const unsigned*)(rowp + 16 * hi);
  const unsigned* p1 = (const unsigned*)(rowp + 16 * hi + 8);
#pragma unroll
  for (int v = 0; v < 4; ++v) { u[v] = p0[v]; u[4 + v] = p1[v]; }
}

// B-fragment (32x16 bf16): lane n = lane&15; v0..7 hold K = 16*hi + {2v,2v+1}
__device__ __forceinline__ void b_frag_load(const __bf16* rowp, int hi, unsigned* u) {
  const unsigned* p = (const unsigned*)(rowp + 16 * hi);
#pragma unroll
  for (int v = 0; v < 8; ++v) u[v] = p[v];
}

__device__ __forceinline__ v8f wmma_bf16(const Frag& a, const Frag& b, v8f c) {
  return __builtin_amdgcn_wmma_f32_16x16x32_bf16(false, a.v, false, b.v,
                                                 (short)0, c, false, false);
}

// CDNA5 async DMA: 16B per lane, global -> LDS, tracked by ASYNCcnt.
// LDS dest operand = wave-relative LDS offset = low 32 bits of a generic
// pointer into __shared__ (ISA 10.2).
__device__ __forceinline__ void async_copy16(const __bf16* g, __bf16* l) {
  unsigned loff = (unsigned)(size_t)l;
  unsigned long long ga = (unsigned long long)(size_t)g;
  asm volatile("global_load_async_to_lds_b128 %0, %1, off"
               :: "v"(loff), "v"(ga) : "memory");
}
__device__ __forceinline__ void wait_async_le4() {
  asm volatile("s_wait_asynccnt 4" ::: "memory");
}
__device__ __forceinline__ void wait_async_0() {
  asm volatile("s_wait_asynccnt 0" ::: "memory");
}

// ---------------- weight cast ----------------
__global__ void cast_f32_bf16(const float* __restrict__ in, __bf16* __restrict__ out, int n) {
  int i = blockIdx.x * blockDim.x + threadIdx.x;
  if (i < n) out[i] = (__bf16)in[i];
}

// ---------------- LayerNorm (f32 -> bf16) ----------------
__global__ __launch_bounds__(256) void ln_kernel(const float* __restrict__ x,
                                                 const float* __restrict__ sc,
                                                 const float* __restrict__ sh,
                                                 __bf16* __restrict__ out) {
  const int row = blockIdx.x;
  const float* xr = x + (size_t)row * DN;
  float s = 0.f, s2 = 0.f;
  for (int i = threadIdx.x; i < DN; i += 256) {
    float v = xr[i]; s += v; s2 += v * v;
  }
#pragma unroll
  for (int off = 16; off; off >>= 1) {
    s  += __shfl_xor(s,  off, 32);
    s2 += __shfl_xor(s2, off, 32);
  }
  __shared__ float red[16];
  const int wave = threadIdx.x >> 5, lane = threadIdx.x & 31;
  if (lane == 0) { red[wave] = s; red[8 + wave] = s2; }
  __syncthreads();
  float ts = 0.f, ts2 = 0.f;
#pragma unroll
  for (int i = 0; i < 8; ++i) { ts += red[i]; ts2 += red[8 + i]; }
  const float mean = ts * (1.0f / DN);
  const float var  = ts2 * (1.0f / DN) - mean * mean;
  const float rstd = rsqrtf(var + 1e-5f);
  for (int i = threadIdx.x; i < DN; i += 256)
    out[(size_t)row * DN + i] = (__bf16)(sc[i] * (xr[i] - mean) * rstd + sh[i]);
}

// ---------------- WMMA GEMM: C[M,N] = A[M,K] * B[N,K]^T ----------------
// Double-buffered async global->LDS staging (4 async b128 ops per wave per
// tile; s_wait_asynccnt 4 => previous tile landed while next is in flight).
enum { MODE_QK = 0, MODE_VT = 1, MODE_F32 = 2, MODE_GELU = 3 };

template <int MODE>
__global__ __launch_bounds__(256) void gemm_bf16_k(
    const __bf16* __restrict__ A, const __bf16* __restrict__ Bw,
    int M, int N, int K,
    const float* __restrict__ bias, const float* __restrict__ resid,
    float* __restrict__ outf, __bf16* __restrict__ outb) {
  __shared__ __align__(16) __bf16 As[2][128 * LDA];
  __shared__ __align__(16) __bf16 Bs[2][128 * LDA];
  const int m0 = blockIdx.y * 128;
  const int n0 = blockIdx.x * 128;
  const int t = threadIdx.x;
  const int lr = t >> 1;            // row this thread stages (0..127)
  const int lc = (t & 1) * 16;      // half-row offset in halves
  const int wave = t >> 5;
  const int lane = t & 31;
  const int ln = lane & 15, hi = lane >> 4;
  const int wm = (wave & 3) * 32;   // 4 waves tile M
  const int wn = (wave >> 2) * 64;  // 2 waves tile N

  const __bf16* gA = A  + (size_t)(m0 + lr) * K + lc;
  const __bf16* gB = Bw + (size_t)(n0 + lr) * K + lc;

  v8f acc[2][4];
#pragma unroll
  for (int i = 0; i < 2; ++i)
#pragma unroll
    for (int j = 0; j < 4; ++j) acc[i][j] = zero8();

  const int nsteps = K >> 5;
  // prologue: stage tile 0 into buffer 0
  {
    __bf16* la = &As[0][lr * LDA + lc];
    __bf16* lb = &Bs[0][lr * LDA + lc];
    async_copy16(gA,     la);  async_copy16(gA + 8, la + 8);
    async_copy16(gB,     lb);  async_copy16(gB + 8, lb + 8);
  }
  for (int s = 0; s < nsteps; ++s) {
    const int cur = s & 1;
    if (s + 1 < nsteps) {
      // safe: trailing barrier of iter s-1 guarantees everyone finished
      // reading buffer (s+1)&1 (used at iter s-1)
      const int nxt = (s + 1) & 1;
      const __bf16* ga = gA + (size_t)(s + 1) * 32;
      const __bf16* gb = gB + (size_t)(s + 1) * 32;
      __bf16* la = &As[nxt][lr * LDA + lc];
      __bf16* lb = &Bs[nxt][lr * LDA + lc];
      async_copy16(ga,     la);  async_copy16(ga + 8, la + 8);
      async_copy16(gb,     lb);  async_copy16(gb + 8, lb + 8);
      wait_async_le4();   // tile s complete (async loads retire in order)
    } else {
      wait_async_0();     // last tile: wait everything
    }
    __syncthreads();      // all waves' tile-s data visible in LDS

    Frag af[2], bf[4];
#pragma unroll
    for (int i = 0; i < 2; ++i) a_frag_load(&As[cur][(wm + i * 16 + ln) * LDA], hi, af[i].u);
#pragma unroll
    for (int j = 0; j < 4; ++j) b_frag_load(&Bs[cur][(wn + j * 16 + ln) * LDA], hi, bf[j].u);
#pragma unroll
    for (int i = 0; i < 2; ++i)
#pragma unroll
      for (int j = 0; j < 4; ++j) acc[i][j] = wmma_bf16(af[i], bf[j], acc[i][j]);

    __syncthreads();      // everyone done reading buffer `cur`
  }

  // Epilogue: C layout: row = g + 8*hi (per 16-tile), col = lane&15
#pragma unroll
  for (int i = 0; i < 2; ++i)
#pragma unroll
    for (int j = 0; j < 4; ++j)
#pragma unroll
      for (int g = 0; g < 8; ++g) {
        const int m = m0 + wm + i * 16 + g + 8 * hi;
        const int n = n0 + wn + j * 16 + ln;
        const float v = acc[i][j][g];
        if (MODE == MODE_QK) {
          const int b = m >> 11, tt = m & (TN - 1);
          const int h = n >> 6,  hd = n & 63;
          outb[(((size_t)(b * HN + h)) * TN + tt) * HDN + hd] = (__bf16)v;
        } else if (MODE == MODE_VT) {
          const int b = m >> 11, tt = m & (TN - 1);
          const int h = n >> 6,  hd = n & 63;
          outb[(((size_t)(b * HN + h)) * HDN + hd) * TN + tt] = (__bf16)v;
        } else if (MODE == MODE_F32) {
          const size_t off = (size_t)m * N + n;
          outf[off] = resid[off] + v + bias[n];
        } else { // MODE_GELU
          const float xv = v + bias[n];
          const float uu = xv + 0.044715f * xv * xv * xv;
          const float gl = 0.5f * xv * (1.0f + tanhf(0.7978845608028654f * uu));
          outb[(size_t)m * N + n] = (__bf16)gl;
        }
      }
}

// ---------------- Flash attention (causal), all-WMMA ----------------
// Computes S^T = K*Q^T so softmax row stats reduce within-lane + one xor-16
// shuffle, then ctx^T = V^T * P^T (V stored transposed [b,h,hd,t]).
__global__ __launch_bounds__(128) void attn_kernel(const __bf16* __restrict__ Qg,
                                                   const __bf16* __restrict__ Kg,
                                                   const __bf16* __restrict__ Vg,
                                                   __bf16* __restrict__ Cg) {
  const int bh = blockIdx.y;
  const int b = bh / HN, h = bh % HN;
  const int wave = threadIdx.x >> 5;
  const int lane = threadIdx.x & 31;
  const int ln = lane & 15, hi = lane >> 4;
  const int qbase = blockIdx.x * 64 + wave * 16;

  const __bf16* Qp = Qg + (size_t)bh * TN * HDN;
  const __bf16* Kp = Kg + (size_t)bh * TN * HDN;
  const __bf16* Vp = Vg + (size_t)bh * HDN * TN;

  // Q as B-fragments (n = q row, K = hd), reused across the whole k-loop
  Frag qb[2];
#pragma unroll
  for (int j = 0; j < 2; ++j)
    b_frag_load(Qp + (size_t)(qbase + ln) * HDN + 32 * j, hi, qb[j].u);

  v8f acc[4];
#pragma unroll
  for (int i = 0; i < 4; ++i) acc[i] = zero8();
  float mprev = -1e30f, lsum = 0.0f;
  const float scale = 0.125f;  // 1/sqrt(64)
  const int qlast = qbase + 15;
  const int qg = qbase + ln;

  for (int ks = 0; ks <= qlast; ks += 32) {
    // S^T tiles (16 tk x 16 q), contraction over hd=64 (2 WMMAs each)
    Frag ka;
    v8f s0 = zero8(), s1 = zero8();
    a_frag_load(Kp + (size_t)(ks + ln) * HDN + 0,  hi, ka.u); s0 = wmma_bf16(ka, qb[0], s0);
    a_frag_load(Kp + (size_t)(ks + ln) * HDN + 32, hi, ka.u); s0 = wmma_bf16(ka, qb[1], s0);
    a_frag_load(Kp + (size_t)(ks + 16 + ln) * HDN + 0,  hi, ka.u); s1 = wmma_bf16(ka, qb[0], s1);
    a_frag_load(Kp + (size_t)(ks + 16 + ln) * HDN + 32, hi, ka.u); s1 = wmma_bf16(ka, qb[1], s1);

    // scale + causal mask (row tk = ks + 16*tile + g + 8*hi; col q = qg)
    float p0[8], p1[8];
    float mx = -1e30f;
#pragma unroll
    for (int g = 0; g < 8; ++g) {
      float v0 = s0[g] * scale;
      float v1 = s1[g] * scale;
      if (ks + g + 8 * hi > qg)       v0 = -1e30f;
      if (ks + 16 + g + 8 * hi > qg)  v1 = -1e30f;
      p0[g] = v0; p1[g] = v1;
      mx = fmaxf(mx, fmaxf(v0, v1));
    }
    mx = fmaxf(mx, __shfl_xor(mx, 16, 32));
    const float mnew = fmaxf(mprev, mx);
    const float alpha = __expf(mprev - mnew);
    float rsum = 0.0f;
#pragma unroll
    for (int g = 0; g < 8; ++g) {
      p0[g] = __expf(p0[g] - mnew);
      p1[g] = __expf(p1[g] - mnew);
      rsum += p0[g] + p1[g];
    }
    rsum += __shfl_xor(rsum, 16, 32);
    lsum = lsum * alpha + rsum;
    mprev = mnew;
#pragma unroll
    for (int i = 0; i < 4; ++i)
#pragma unroll
      for (int g = 0; g < 8; ++g) acc[i][g] *= alpha;

    // Transpose P (C layout) into a B-fragment of P^T via one xor-16 shuffle
    float sx0[8], sx1[8];
#pragma unroll
    for (int g = 0; g < 8; ++g) {
      sx0[g] = __shfl_xor(p0[g], 16, 32);
      sx1[g] = __shfl_xor(p1[g], 16, 32);
    }
    Frag pb;
#pragma unroll
    for (int v = 0; v < 4; ++v) {
      const float a0 = hi ? sx1[2 * v]     : p0[2 * v];
      const float a1 = hi ? sx1[2 * v + 1] : p0[2 * v + 1];
      pb.u[v] = pack2(a0, a1);
    }
#pragma unroll
    for (int v = 0; v < 4; ++v) {
      const float a0 = hi ? p1[2 * v]     : sx0[2 * v];
      const float a1 = hi ? p1[2 * v + 1] : sx0[2 * v + 1];
      pb.u[4 + v] = pack2(a0, a1);
    }

    // ctx^T += V^T (A-frag, rows = hd, contiguous over t) * P^T (B-frag)
    Frag va;
#pragma unroll
    for (int i = 0; i < 4; ++i) {
      a_frag_load(Vp + (size_t)(16 * i + ln) * TN + ks, hi, va.u);
      acc[i] = wmma_bf16(va, pb, acc[i]);
    }
  }

  const float linv = 1.0f / lsum;  // each q row always has >=1 unmasked key
#pragma unroll
  for (int i = 0; i < 4; ++i)
#pragma unroll
    for (int g = 0; g < 8; ++g) {
      const int hd = 16 * i + g + 8 * hi;
      Cg[((size_t)(b * TN + qbase + ln)) * DN + h * HDN + hd] = (__bf16)(acc[i][g] * linv);
    }
}

// ---------------- host launcher ----------------
extern "C" void kernel_launch(void* const* d_in, const int* in_sizes, int n_in,
                              void* d_out, int out_size, void* d_ws, size_t ws_size,
                              hipStream_t stream) {
  (void)in_sizes; (void)n_in; (void)out_size; (void)ws_size;
  const float* x    = (const float*)d_in[0];
  const float* ln1s = (const float*)d_in[1];
  const float* ln1b = (const float*)d_in[2];
  const float* wq   = (const float*)d_in[3];
  const float* wk   = (const float*)d_in[4];
  const float* wv   = (const float*)d_in[5];
  const float* wo   = (const float*)d_in[6];
  const float* bo   = (const float*)d_in[7];
  const float* ln2s = (const float*)d_in[8];
  const float* ln2b = (const float*)d_in[9];
  const float* w1   = (const float*)d_in[10];
  const float* b1   = (const float*)d_in[11];
  const float* w2   = (const float*)d_in[12];
  const float* b2   = (const float*)d_in[13];
  float* out = (float*)d_out;

  // Workspace carve-up (~58 MB)
  __bf16* XN  = (__bf16*)d_ws;                 // [BT, DN] bf16 (xn1, later xn2)
  __bf16* WQb = XN  + (size_t)BT * DN;
  __bf16* WKb = WQb + (size_t)DN * DN;
  __bf16* WVb = WKb + (size_t)DN * DN;
  __bf16* WOb = WVb + (size_t)DN * DN;
  __bf16* W1b = WOb + (size_t)DN * DN;         // [D4, DN]
  __bf16* W2b = W1b + (size_t)D4 * DN;         // [DN, D4]
  float*  X1  = (float*)(W2b + (size_t)DN * D4);   // [BT, DN] f32 residual
  __bf16* Qb  = (__bf16*)(X1 + (size_t)BT * DN);   // [b,h,t,hd]
  __bf16* Kb  = Qb + (size_t)BT * DN;              // [b,h,t,hd]
  __bf16* Vb  = Kb + (size_t)BT * DN;              // [b,h,hd,t]  (transposed)
  __bf16* Cb  = Vb + (size_t)BT * DN;              // ctx [b,t,D] bf16
  __bf16* Hb  = Qb;  // MLP hidden [BT, D4] aliases q/k/v/ctx (consumed by then)

  const int nD2 = DN * DN, nW1 = D4 * DN;
  cast_f32_bf16<<<(nD2 + 255) / 256, 256, 0, stream>>>(wq, WQb, nD2);
  cast_f32_bf16<<<(nD2 + 255) / 256, 256, 0, stream>>>(wk, WKb, nD2);
  cast_f32_bf16<<<(nD2 + 255) / 256, 256, 0, stream>>>(wv, WVb, nD2);
  cast_f32_bf16<<<(nD2 + 255) / 256, 256, 0, stream>>>(wo, WOb, nD2);
  cast_f32_bf16<<<(nW1 + 255) / 256, 256, 0, stream>>>(w1, W1b, nW1);
  cast_f32_bf16<<<(nW1 + 255) / 256, 256, 0, stream>>>(w2, W2b, nW1);

  ln_kernel<<<BT, 256, 0, stream>>>(x, ln1s, ln1b, XN);

  const dim3 gD(DN / 128, BT / 128);   // (6, 32)
  const dim3 gH(D4 / 128, BT / 128);   // (24, 32)
  gemm_bf16_k<MODE_QK><<<gD, 256, 0, stream>>>(XN, WQb, BT, DN, DN, nullptr, nullptr, nullptr, Qb);
  gemm_bf16_k<MODE_QK><<<gD, 256, 0, stream>>>(XN, WKb, BT, DN, DN, nullptr, nullptr, nullptr, Kb);
  gemm_bf16_k<MODE_VT><<<gD, 256, 0, stream>>>(XN, WVb, BT, DN, DN, nullptr, nullptr, nullptr, Vb);

  attn_kernel<<<dim3(TN / 64, BN * HN), 128, 0, stream>>>(Qb, Kb, Vb, Cb);

  gemm_bf16_k<MODE_F32><<<gD, 256, 0, stream>>>(Cb, WOb, BT, DN, DN, bo, x, X1, nullptr);
  ln_kernel<<<BT, 256, 0, stream>>>(X1, ln2s, ln2b, XN);
  gemm_bf16_k<MODE_GELU><<<gH, 256, 0, stream>>>(XN, W1b, BT, D4, DN, b1, nullptr, nullptr, Hb);
  gemm_bf16_k<MODE_F32><<<gD, 256, 0, stream>>>(Hb, W2b, BT, DN, D4, b2, X1, out, nullptr);
}
